// TorchMD_Net_8117488189528
// MI455X (gfx1250) — compile-verified, matrix operand order checked by reading, not run on hardware
//
#include <hip/hip_runtime.h>
#include <hip/hip_bf16.h>

#define NUM_GRAPHS 2048
#define MAX_NODES  128
#define HIDDEN     256
#define OUT_CH     12

typedef __attribute__((ext_vector_type(16))) __bf16 v16bf;
typedef __attribute__((ext_vector_type(8)))  __bf16 v8bf;
typedef __attribute__((ext_vector_type(8)))  float  v8f;

union Frag16 { v16bf v; v8bf h[2]; };

// ---- LDS layout (bytes), dynamic shared memory ----
// sA    : 128 x 136 bf16  (adjacency, row-major [m][k])       34816 B
// sXT   : 256 x 136 bf16  (X transposed [h][m])               69632 B
// sW0   : 256 x 136 bf16  (walk^T buffer 0)                   69632 B
// sW1   : 256 x 136 bf16  (walk^T buffer 1)                   69632 B
// vacc  : 6 x 256 f32     (v1..v6 column sums)                 6144 B
#define LDA 136
#define LDT 136
#define OFF_A   0
#define OFF_XT  (OFF_A  + 128*LDA*2)
#define OFF_W0  (OFF_XT + 256*LDT*2)
#define OFF_W1  (OFF_W0 + 256*LDT*2)
#define OFF_V   (OFF_W1 + 256*LDT*2)
#define SMEM_BYTES (OFF_V + 6*256*4)   // 249856 B < 320 KB WGP LDS

__global__ void zero_words(unsigned* p, int n) {
    int i = blockIdx.x * blockDim.x + threadIdx.x;
    if (i < n) p[i] = 0u;
}

// Scatter edges into per-graph 128x128 bitmask: word layout [g][row][4]
__global__ void build_adj_bits(const int* __restrict__ ei, unsigned* __restrict__ abits, int nE) {
    int e = blockIdx.x * blockDim.x + threadIdx.x;
    if (e >= nE) return;
    int s = ei[e];
    int d = ei[nE + e];
    int g  = s >> 7;
    int li = s & 127;
    int lj = d & 127;
    atomicOr(&abits[(g * 128 + li) * 4 + (lj >> 5)], 1u << (lj & 31));
    atomicOr(&abits[(g * 128 + lj) * 4 + (li >> 5)], 1u << (li & 31));
}

__global__ __launch_bounds__(256, 1)
void ngram_walk_kernel(const float* __restrict__ x,
                       const float* __restrict__ Wp,
                       const float* __restrict__ bp,
                       const unsigned* __restrict__ abits,
                       float* __restrict__ out) {
    extern __shared__ char smem[];
    __bf16* sA   = (__bf16*)(smem + OFF_A);
    __bf16* sXT  = (__bf16*)(smem + OFF_XT);
    __bf16* sW0  = (__bf16*)(smem + OFF_W0);
    __bf16* sW1  = (__bf16*)(smem + OFF_W1);
    float*  vacc = (float*) (smem + OFF_V);

    const int g   = blockIdx.x;
    const int tid = threadIdx.x;

    // ---- expand adjacency bitmask -> bf16 tile in LDS ----
    const unsigned* ab = abits + (size_t)g * 512;
    for (int w = tid; w < 512; w += 256) {
        unsigned bits = ab[w];
        int m  = w >> 2;
        int kb = (w & 3) * 32;
        #pragma unroll 8
        for (int bpos = 0; bpos < 32; ++bpos)
            sA[m * LDA + kb + bpos] = ((bits >> bpos) & 1u) ? (__bf16)1.0f : (__bf16)0.0f;
    }

    // ---- load X -> XT (bf16, transposed), init walk buffer 0, compute v1 ----
    // thread tid owns column h = tid; coalesced global reads across threads.
    {
        const float* xg = x + (size_t)g * MAX_NODES * HIDDEN;
        float v1s = 0.0f;
        for (int m = 0; m < MAX_NODES; ++m) {
            float val = xg[m * HIDDEN + tid];
            __bf16 bv = (__bf16)val;
            sXT[tid * LDT + m] = bv;
            sW0[tid * LDT + m] = bv;
            v1s += val;
        }
        vacc[tid] = v1s;                       // v1
        #pragma unroll
        for (int k = 1; k < 6; ++k) vacc[k * 256 + tid] = 0.0f;
    }
    __syncthreads();

    // ---- 5 walk iterations: walk = (A @ walk) * X, sums into vacc ----
    const int wv    = tid >> 5;       // wave id 0..7
    const int lane  = tid & 31;
    const int khalf = lane >> 4;      // 0: lanes 0-15, 1: lanes 16-31
    const int mloc  = lane & 15;      // M index (A) / N index (B) within tile
    const int rowbase = wv * 16;      // output row tile for this wave

    for (int iter = 1; iter <= 5; ++iter) {
        const __bf16* cur = (iter & 1) ? sW0 : sW1;
        __bf16*       nxt = (iter & 1) ? sW1 : sW0;

        // A-operand fragments for this wave's 16 rows, all 4 K-tiles (reused 16x)
        Frag16 af[4];
        const int arow = rowbase + mloc;
        #pragma unroll
        for (int kt = 0; kt < 4; ++kt) {
            int k0 = kt * 32 + khalf * 8;
            af[kt].h[0] = *(const v8bf*)(sA + arow * LDA + k0);
            af[kt].h[1] = *(const v8bf*)(sA + arow * LDA + k0 + 16);
        }

        for (int ht = 0; ht < 16; ++ht) {
            const int hrow = ht * 16 + mloc;   // h column handled by this lane
            v8f acc = {};
            #pragma unroll
            for (int kt = 0; kt < 4; ++kt) {
                Frag16 bfg;
                int k0 = kt * 32 + khalf * 8;
                bfg.h[0] = *(const v8bf*)(cur + hrow * LDT + k0);
                bfg.h[1] = *(const v8bf*)(cur + hrow * LDT + k0 + 16);
                acc = __builtin_amdgcn_wmma_f32_16x16x32_bf16(
                        false, af[kt].v, false, bfg.v, (short)0, acc, false, false);
            }
            // elementwise * X, convert to bf16, store to next walk buffer,
            // accumulate column sum
            const int m0 = rowbase + khalf * 8;           // 8 contiguous node rows
            v8bf xv = *(const v8bf*)(sXT + hrow * LDT + m0);
            v8bf ov;
            float s = 0.0f;
            #pragma unroll
            for (int r = 0; r < 8; ++r) {
                float nv = acc[r] * (float)xv[r];
                s += nv;
                ov[r] = (__bf16)nv;
            }
            *(v8bf*)(nxt + hrow * LDT + m0) = ov;
            atomicAdd(&vacc[iter * 256 + hrow], s);       // ds_add_f32
        }
        __syncthreads();
    }

    // ---- projection: out[g, c] = b[c] + sum_j vacc[j] * W[c, j] ----
    if (tid < OUT_CH) {
        const float* wr = Wp + tid * (6 * HIDDEN);
        float s = bp[tid];
        for (int j = 0; j < 6 * HIDDEN; ++j) s += vacc[j] * wr[j];
        out[(size_t)g * OUT_CH + tid] = s;
    }
}

extern "C" void kernel_launch(void* const* d_in, const int* in_sizes, int n_in,
                              void* d_out, int out_size, void* d_ws, size_t ws_size,
                              hipStream_t stream) {
    const float* x  = (const float*)d_in[0];
    const float* W  = (const float*)d_in[1];
    const float* b  = (const float*)d_in[2];
    // d_in[3] = batch (uniform: node i -> graph i>>7), not needed
    const int*   ei = (const int*)d_in[4];
    float* out = (float*)d_out;

    const int nE = in_sizes[4] / 2;
    unsigned* abits = (unsigned*)d_ws;           // 2048*128*4 words = 4 MB
    const int nwords = NUM_GRAPHS * MAX_NODES * 4;

    zero_words<<<(nwords + 255) / 256, 256, 0, stream>>>(abits, nwords);
    build_adj_bits<<<(nE + 255) / 256, 256, 0, stream>>>(ei, abits, nE);
    ngram_walk_kernel<<<NUM_GRAPHS, 256, SMEM_BYTES, stream>>>(x, W, b, abits, out);
}
